// STTR_block_48902497632569
// MI455X (gfx1250) — matrix operand order repaired and checked by reading, not compiled, not run on hardware
//
#include <hip/hip_runtime.h>
#include <math.h>

// ---- problem constants (from reference) ----
constexpr int kN = 16, kB = 8, kT = 64, kC = 256, kH = 8, kF = 1024, kK = 8;
constexpr int kS = kT * kN;      // 1024
constexpr int kD = kC / kH;      // 32
constexpr float kScale = 0.17677669529663687f; // 1/sqrt(32)

typedef __attribute__((ext_vector_type(16))) _Float16 v16h;
typedef __attribute__((ext_vector_type(8)))  _Float16 v8h;
typedef __attribute__((ext_vector_type(4)))  _Float16 v4h;
typedef __attribute__((ext_vector_type(8)))  float    v8f;

// ---------------- wave reduction helpers (wave32) ----------------
__device__ __forceinline__ float wave_sum(float v) {
#pragma unroll
  for (int o = 16; o > 0; o >>= 1) v += __shfl_xor(v, o, 32);
  return v;
}
__device__ __forceinline__ float wave_max(float v) {
#pragma unroll
  for (int o = 16; o > 0; o >>= 1) v = fmaxf(v, __shfl_xor(v, o, 32));
  return v;
}

// ---------------- f32 -> f16 bulk convert (vectorized, n % 4 == 0 in all uses) ----
__global__ void f32_to_f16_k(const float* __restrict__ src, _Float16* __restrict__ dst, int n)
{
  const int i4 = (blockIdx.x * blockDim.x + threadIdx.x) * 4;
  if (i4 >= n) return;
  const float4 v = *(const float4*)(src + i4);
  v4h h;
  h[0] = (_Float16)v.x; h[1] = (_Float16)v.y;
  h[2] = (_Float16)v.z; h[3] = (_Float16)v.w;
  *(v4h*)(dst + i4) = h;
}

// ---------------- WMMA GEMM:  out[M,N] = A[M,K] @ W[N,K]^T + bias (+relu) (+residual)
// A, W pre-converted to f16. One 16x64 tile per wave32: 4 accumulators share one
// A fragment; K stepped by 32 via v_wmma_f32_16x16x32_f16. Fragments are manually
// double-buffered across K-steps so the loads for step k+1 are in flight while the
// WMMAs for step k execute (partial s_wait_loadcnt instead of full drains).
// M-tail handled by clamping the A row (dead outputs are never stored).
__global__ void gemm_f16_wmma(const _Float16* __restrict__ A,
                              const _Float16* __restrict__ W,
                              const float* __restrict__ bias,
                              const float* __restrict__ resid,
                              float* __restrict__ out,
                              int M, int Nc, int Kc, int doRelu)
{
  const int lane  = threadIdx.x & 31;
  const int wave  = blockIdx.x * (blockDim.x >> 5) + (threadIdx.x >> 5);
  const int tilesN = Nc >> 6;                 // 64 columns per wave
  const int tilesM = (M + 15) >> 4;
  if (wave >= tilesM * tilesN) return;        // wave-uniform exit
  const int tm = wave / tilesN;
  const int tn = wave - tm * tilesN;

  const int l15 = lane & 15;
  int arow = tm * 16 + l15;
  if (arow >= M) arow = M - 1;                // clamp: extra rows are never stored
  const _Float16* __restrict__ Ap  = A + (size_t)arow * Kc;
  const _Float16* __restrict__ Wp0 = W + (size_t)(tn * 64 + l15) * Kc;
  const _Float16* __restrict__ Wp1 = Wp0 + (size_t)16 * Kc;
  const _Float16* __restrict__ Wp2 = Wp0 + (size_t)32 * Kc;
  const _Float16* __restrict__ Wp3 = Wp0 + (size_t)48 * Kc;
  const int kA  = (lane < 16) ? 0 : 8;        // A 16x32 f16 lane layout (ISA 7.12.2)
  const int kB_ = (lane < 16) ? 0 : 8;        // B 32x16 f16 lane layout

  auto loadA = [&](int k0) -> v16h {
    const v8h lo = *(const v8h*)(Ap + k0 + kA);
    const v8h hi = *(const v8h*)(Ap + k0 + kA + 16);
    return __builtin_shufflevector(lo, hi, 0,1,2,3,4,5,6,7,8,9,10,11,12,13,14,15);
  };
  auto loadB = [&](const _Float16* __restrict__ Wp, int k0) -> v16h {
    const v8h lo = *(const v8h*)(Wp + k0 + kB_);
    const v8h hi = *(const v8h*)(Wp + k0 + kB_ + 16);
    return __builtin_shufflevector(lo, hi, 0,1,2,3,4,5,6,7,8,9,10,11,12,13,14,15);
  };

  v8f acc0 = {0.f,0.f,0.f,0.f,0.f,0.f,0.f,0.f};
  v8f acc1 = acc0, acc2 = acc0, acc3 = acc0;

  // prologue: fragments for k=0
  v16h av  = loadA(0);
  v16h bv0 = loadB(Wp0, 0), bv1 = loadB(Wp1, 0);
  v16h bv2 = loadB(Wp2, 0), bv3 = loadB(Wp3, 0);

  for (int k0 = 0; k0 < Kc; k0 += 32) {
    const int kn = (k0 + 32 < Kc) ? (k0 + 32) : k0;   // last iter: dead reload
    // issue next-step loads before consuming current fragments
    const v16h avN  = loadA(kn);
    const v16h b0N  = loadB(Wp0, kn);
    const v16h b1N  = loadB(Wp1, kn);
    const v16h b2N  = loadB(Wp2, kn);
    const v16h b3N  = loadB(Wp3, kn);
    acc0 = __builtin_amdgcn_wmma_f32_16x16x32_f16(false, av, false, bv0, (short)0, acc0, false, false);
    acc1 = __builtin_amdgcn_wmma_f32_16x16x32_f16(false, av, false, bv1, (short)0, acc1, false, false);
    acc2 = __builtin_amdgcn_wmma_f32_16x16x32_f16(false, av, false, bv2, (short)0, acc2, false, false);
    acc3 = __builtin_amdgcn_wmma_f32_16x16x32_f16(false, av, false, bv3, (short)0, acc3, false, false);
    av = avN; bv0 = b0N; bv1 = b1N; bv2 = b2N; bv3 = b3N;
  }

  const int col0 = tn * 64 + l15;
  const float bb0 = bias[col0];
  const float bb1 = bias[col0 + 16];
  const float bb2 = bias[col0 + 32];
  const float bb3 = bias[col0 + 48];
  const int mbase = tm * 16 + ((lane < 16) ? 0 : 8);  // D layout: VGPR v -> row v (+8 hi lanes)

  if (tm * 16 + 16 <= M) {                    // uniform fast path: full tile in bounds
#pragma unroll
    for (int v = 0; v < 8; ++v) {
      const size_t m = (size_t)(mbase + v);
      float r0 = acc0[v] + bb0, r1 = acc1[v] + bb1;
      float r2 = acc2[v] + bb2, r3 = acc3[v] + bb3;
      if (doRelu) {
        r0 = fmaxf(r0, 0.f); r1 = fmaxf(r1, 0.f);
        r2 = fmaxf(r2, 0.f); r3 = fmaxf(r3, 0.f);
      }
      if (resid) {
        r0 += resid[m * Nc + col0];      r1 += resid[m * Nc + col0 + 16];
        r2 += resid[m * Nc + col0 + 32]; r3 += resid[m * Nc + col0 + 48];
      }
      out[m * Nc + col0]      = r0; out[m * Nc + col0 + 16] = r1;
      out[m * Nc + col0 + 32] = r2; out[m * Nc + col0 + 48] = r3;
    }
  } else {                                    // M-tail tile (only the M=8 GEMM)
#pragma unroll
    for (int v = 0; v < 8; ++v) {
      const int m = mbase + v;
      if (m < M) {
        float r0 = acc0[v] + bb0, r1 = acc1[v] + bb1;
        float r2 = acc2[v] + bb2, r3 = acc3[v] + bb3;
        if (doRelu) {
          r0 = fmaxf(r0, 0.f); r1 = fmaxf(r1, 0.f);
          r2 = fmaxf(r2, 0.f); r3 = fmaxf(r3, 0.f);
        }
        if (resid) {
          r0 += resid[(size_t)m * Nc + col0];      r1 += resid[(size_t)m * Nc + col0 + 16];
          r2 += resid[(size_t)m * Nc + col0 + 32]; r3 += resid[(size_t)m * Nc + col0 + 48];
        }
        out[(size_t)m * Nc + col0]      = r0; out[(size_t)m * Nc + col0 + 16] = r1;
        out[(size_t)m * Nc + col0 + 32] = r2; out[(size_t)m * Nc + col0 + 48] = r3;
      }
    }
  }
}

// ---------------- xs = x.transpose(2,0,1,3).reshape(T*N, B, C) ----------------
__global__ void build_xs_k(const float* __restrict__ x, float* __restrict__ xs)
{
  int i = blockIdx.x * blockDim.x + threadIdx.x;
  const int total = kS * kB * kC;
  if (i >= total) return;
  const int c = i % kC; int r = i / kC;
  const int b = r % kB; const int s = r / kB;
  const int t = s / kN, n = s % kN;
  xs[i] = x[(((size_t)n * kB + b) * kT + t) * kC + c];
}

// ---------------- k-NN (8 nearest incl self) from box [B,S,2] ----------------
__global__ void knn_k(const float* __restrict__ box, int* __restrict__ idx)
{
  int i = blockIdx.x * blockDim.x + threadIdx.x;
  if (i >= kB * kS) return;
  const int b = i / kS, s = i % kS;
  const float qx = box[((size_t)b * kS + s) * 2 + 0];
  const float qy = box[((size_t)b * kS + s) * 2 + 1];
  float bd[kK]; int bi[kK];
#pragma unroll
  for (int j = 0; j < kK; ++j) { bd[j] = 3.4e38f; bi[j] = 0; }
  for (int j = 0; j < kS; ++j) {
    const float dx = box[((size_t)b * kS + j) * 2 + 0] - qx;
    const float dy = box[((size_t)b * kS + j) * 2 + 1] - qy;
    const float d2 = dx * dx + dy * dy;
    if (d2 < bd[kK - 1]) {
      int p = kK - 1;
      while (p > 0 && bd[p - 1] > d2) { bd[p] = bd[p - 1]; bi[p] = bi[p - 1]; --p; }
      bd[p] = d2; bi[p] = j;
    }
  }
#pragma unroll
  for (int j = 0; j < kK; ++j) idx[(size_t)i * kK + j] = bi[j];
}

// ---------------- encoder gather attention (exactly equals masked softmax) ----------------
// qkv: [S*B, 768] (q|k|v), out: [S*B, 256]
__global__ void enc_attn_k(const float* __restrict__ qkv, const int* __restrict__ idx,
                           float* __restrict__ out)
{
  int i = blockIdx.x * blockDim.x + threadIdx.x;
  if (i >= kB * kH * kS) return;
  const int qi = i % kS; int r = i / kS;
  const int h = r % kH;  const int b = r / kH;
  const float* qp = qkv + ((size_t)qi * kB + b) * (3 * kC) + h * kD;
  int kj[kK]; float sc[kK];
  float mx = -3.4e38f;
#pragma unroll
  for (int j = 0; j < kK; ++j) {
    kj[j] = idx[((size_t)b * kS + qi) * kK + j];
    const float* kp = qkv + ((size_t)kj[j] * kB + b) * (3 * kC) + kC + h * kD;
    float s = 0.f;
#pragma unroll
    for (int d = 0; d < kD; ++d) s += qp[d] * kp[d];
    s *= kScale;
    sc[j] = s; mx = fmaxf(mx, s);
  }
  float den = 0.f;
#pragma unroll
  for (int j = 0; j < kK; ++j) { sc[j] = __expf(sc[j] - mx); den += sc[j]; }
  const float inv = 1.f / den;
  float o[kD];
#pragma unroll
  for (int d = 0; d < kD; ++d) o[d] = 0.f;
#pragma unroll
  for (int j = 0; j < kK; ++j) {
    const float w = sc[j] * inv;
    const float* vp = qkv + ((size_t)kj[j] * kB + b) * (3 * kC) + 2 * kC + h * kD;
#pragma unroll
    for (int d = 0; d < kD; ++d) o[d] += w * vp[d];
  }
  float* op = out + ((size_t)qi * kB + b) * kC + h * kD;
#pragma unroll
  for (int d = 0; d < kD; ++d) op[d] = o[d];
}

// ---------------- LayerNorm over C=256, one wave per row (in-place safe) ----------------
__global__ void layernorm_k(const float* __restrict__ src, float* __restrict__ dst,
                            const float* __restrict__ g, const float* __restrict__ b,
                            int rows)
{
  const int lane = threadIdx.x & 31;
  const int row  = blockIdx.x * (blockDim.x >> 5) + (threadIdx.x >> 5);
  if (row >= rows) return;
  const float* p = src + (size_t)row * kC;
  float v[8]; float s = 0.f;
#pragma unroll
  for (int e = 0; e < 8; ++e) { v[e] = p[e * 32 + lane]; s += v[e]; }
  const float mean = wave_sum(s) * (1.f / kC);
  float q = 0.f;
#pragma unroll
  for (int e = 0; e < 8; ++e) { const float d = v[e] - mean; q += d * d; }
  const float inv = rsqrtf(wave_sum(q) * (1.f / kC) + 1e-5f);
  float* o = dst + (size_t)row * kC;
#pragma unroll
  for (int e = 0; e < 8; ++e) {
    const int c = e * 32 + lane;
    o[c] = (v[e] - mean) * inv * g[c] + b[c];
  }
}

// ---------------- dense attention pieces (decoders) ----------------
__global__ void scores_k(const float* __restrict__ q, int ldq,
                         const float* __restrict__ kv, int ldkv, int ok,
                         float* __restrict__ sc, int Lq, int Lk)
{
  long long i = (long long)blockIdx.x * blockDim.x + threadIdx.x;
  const long long total = (long long)kB * kH * Lq * Lk;
  if (i >= total) return;
  const int ki = (int)(i % Lk); long long r = i / Lk;
  const int qi = (int)(r % Lq); r /= Lq;
  const int h  = (int)(r % kH);
  const int b  = (int)(r / kH);
  const float* qp = q  + ((size_t)qi * kB + b) * ldq + h * kD;
  const float* kp = kv + ((size_t)ki * kB + b) * ldkv + ok + h * kD;
  float s = 0.f;
#pragma unroll
  for (int d = 0; d < kD; ++d) s += qp[d] * kp[d];
  sc[i] = s * kScale;
}

__global__ void softmax_rows_k(float* __restrict__ sc, int rows, int L)
{
  const int lane = threadIdx.x & 31;
  const int row  = blockIdx.x * (blockDim.x >> 5) + (threadIdx.x >> 5);
  if (row >= rows) return;
  float* p = sc + (size_t)row * L;
  float mx = -3.4e38f;
  for (int j = lane; j < L; j += 32) mx = fmaxf(mx, p[j]);
  mx = wave_max(mx);
  float s = 0.f;
  for (int j = lane; j < L; j += 32) { const float e = __expf(p[j] - mx); p[j] = e; s += e; }
  s = wave_sum(s);
  const float inv = 1.f / s;
  for (int j = lane; j < L; j += 32) p[j] *= inv;
}

__global__ void av_k(const float* __restrict__ sc, const float* __restrict__ kv,
                     int ldkv, int ov, float* __restrict__ out, int Lq, int Lk)
{
  int i = blockIdx.x * blockDim.x + threadIdx.x;
  const int total = kB * kH * Lq * kD;
  if (i >= total) return;
  const int d = i % kD; int r = i / kD;
  const int qi = r % Lq; r /= Lq;
  const int h = r % kH;  const int b = r / kH;
  const float* sp = sc + (size_t)((b * kH + h) * Lq + qi) * Lk;
  float acc = 0.f;
  for (int ki = 0; ki < Lk; ++ki)
    acc += sp[ki] * kv[((size_t)ki * kB + b) * ldkv + ov + h * kD + d];
  out[((size_t)qi * kB + b) * kC + h * kD + d] = acc;
}

// ---------------- means ----------------
__global__ void mean_over_t_k(const float* __restrict__ xs, float* __restrict__ out)
{
  int i = blockIdx.x * blockDim.x + threadIdx.x;
  const int total = kN * kB * kC;
  if (i >= total) return;
  const int c = i % kC; int r = i / kC;
  const int b = r % kB; const int n = r / kB;
  float s = 0.f;
  for (int t = 0; t < kT; ++t)
    s += xs[((size_t)(t * kN + n) * kB + b) * kC + c];
  out[i] = s * (1.f / kT);
}

__global__ void mean_rows_k(const float* __restrict__ src, float* __restrict__ out, int L)
{
  int i = blockIdx.x * blockDim.x + threadIdx.x;
  if (i >= kB * kC) return;
  const int c = i % kC; const int b = i / kC;
  float s = 0.f;
  for (int l = 0; l < L; ++l) s += src[((size_t)l * kB + b) * kC + c];
  out[i] = s / (float)L;
}

// ---------------- host-side helpers ----------------
static inline void cvt_launch(const float* src, _Float16* dst, size_t n, hipStream_t s)
{
  f32_to_f16_k<<<(int)((n / 4 + 255) / 256), 256, 0, s>>>(src, dst, (int)n);
}
// A (f32) is converted once to f16 scratch, then the WMMA GEMM re-reads it cheaply.
static inline void gemm_launch(const float* A32, _Float16* a16, const _Float16* W,
                               const float* b, const float* res, float* out,
                               int M, int Nc, int Kc, int relu, hipStream_t s)
{
  cvt_launch(A32, a16, (size_t)M * Kc, s);
  const int waves  = ((M + 15) / 16) * (Nc / 64);      // 16x64 tile per wave
  const int blocks = (waves + 7) / 8;                  // 8 waves (256 threads) per block
  gemm_f16_wmma<<<blocks, 256, 0, s>>>(a16, W, b, res, out, M, Nc, Kc, relu);
}
static inline void ln_launch(const float* src, float* dst, const float* g,
                             const float* b, int rows, hipStream_t s)
{
  layernorm_k<<<(rows + 7) / 8, 256, 0, s>>>(src, dst, g, b, rows);
}
static inline void softmax_launch(float* sc, int rows, int L, hipStream_t s)
{
  softmax_rows_k<<<(rows + 7) / 8, 256, 0, s>>>(sc, rows, L);
}

struct EncP {
  const _Float16 *Wqkv, *Wo, *W1, *W2;
  const float *bqkv, *bo, *b1, *b2, *n1g, *n1b, *n2g, *n2b, *nfg, *nfb;
};
struct DecP {
  const _Float16 *saW, *saWo, *caW, *caWo, *W1, *W2;
  const float *sab, *sabo, *cab, *cabo, *b1, *b2,
              *n1g, *n1b, *n2g, *n2b, *n3g, *n3b, *nfg, *nfb;
};

static void run_decoder(float* tgt, int Lq, const float* mem, int Lk, const DecP& P,
                        float* finalDst, float* t2, float* qkv, float* qb,
                        float* kvb, float* sc, float* att, float* hid,
                        _Float16* a16, hipStream_t s)
{
  const int rq = Lq * kB, rk = Lk * kB;
  // --- self attention (pre-norm) ---
  ln_launch(tgt, t2, P.n1g, P.n1b, rq, s);
  gemm_launch(t2, a16, P.saW, P.sab, nullptr, qkv, rq, 3 * kC, kC, 0, s);
  {
    const long long tot = (long long)kB * kH * Lq * Lq;
    scores_k<<<(int)((tot + 255) / 256), 256, 0, s>>>(qkv, 3 * kC, qkv, 3 * kC, kC, sc, Lq, Lq);
    softmax_launch(sc, kB * kH * Lq, Lq, s);
    const int tot2 = kB * kH * Lq * kD;
    av_k<<<(tot2 + 255) / 256, 256, 0, s>>>(sc, qkv, 3 * kC, 2 * kC, att, Lq, Lq);
  }
  gemm_launch(att, a16, P.saWo, P.sabo, tgt, tgt, rq, kC, kC, 0, s);
  // --- cross attention ---
  ln_launch(tgt, t2, P.n2g, P.n2b, rq, s);
  gemm_launch(t2, a16, P.caW, P.cab, nullptr, qb, rq, kC, kC, 0, s);
  gemm_launch(mem, a16, P.caW + (size_t)kC * kC, P.cab + kC, nullptr, kvb, rk, 2 * kC, kC, 0, s);
  {
    const long long tot = (long long)kB * kH * Lq * Lk;
    scores_k<<<(int)((tot + 255) / 256), 256, 0, s>>>(qb, kC, kvb, 2 * kC, 0, sc, Lq, Lk);
    softmax_launch(sc, kB * kH * Lq, Lk, s);
    const int tot2 = kB * kH * Lq * kD;
    av_k<<<(tot2 + 255) / 256, 256, 0, s>>>(sc, kvb, 2 * kC, kC, att, Lq, Lk);
  }
  gemm_launch(att, a16, P.caWo, P.cabo, tgt, tgt, rq, kC, kC, 0, s);
  // --- FFN ---
  ln_launch(tgt, t2, P.n3g, P.n3b, rq, s);
  gemm_launch(t2, a16, P.W1, P.b1, nullptr, hid, rq, kF, kC, 1, s);
  gemm_launch(hid, a16, P.W2, P.b2, tgt, tgt, rq, kC, kF, 0, s);
  // --- final norm of the stack ---
  ln_launch(tgt, finalDst, P.nfg, P.nfb, rq, s);
}

extern "C" void kernel_launch(void* const* d_in, const int* in_sizes, int n_in,
                              void* d_out, int out_size, void* d_ws, size_t ws_size,
                              hipStream_t stream)
{
  (void)in_sizes; (void)n_in; (void)out_size; (void)ws_size;
  auto f = [&](int i) { return (const float*)d_in[i]; };
  const float* x     = f(0);
  const float* token = f(1);
  const float* box   = f(2);

  // ---- workspace carve (bytes) ----
  char* base = (char*)d_ws;
  size_t o = 0;
  auto takeF = [&](size_t n) { float* p = (float*)(base + o);
                               o += ((n * 4 + 255) & ~(size_t)255); return p; };
  auto takeH = [&](size_t n) { _Float16* p = (_Float16*)(base + o);
                               o += ((n * 2 + 255) & ~(size_t)255); return p; };
  float* bufX = takeF((size_t)kS * kB * kC);        // xs (current residual stream)
  float* bufC = takeF((size_t)kS * kB * kC);        // LN output / tmp
  float* bufD = takeF((size_t)kS * kB * kC);        // attention output
  float* bufB = takeF((size_t)kS * kB * 3 * kC);    // enc qkv / decoder cross-KV
  float* bufA = takeF((size_t)kS * kB * kF);        // enc FFN hidden / scores(+dec hidden)
  float* outN = takeF((size_t)kN * kB * kC);        // encoder mean -> dec_n tgt
  float* tokB = takeF((size_t)kT * kB * kC);        // dec_st tgt (from token)
  float* tbuf = takeF((size_t)kB * kC);             // dec_t tgt
  float* t2b  = takeF((size_t)kT * kB * kC);
  float* qkvq = takeF((size_t)kT * kB * 3 * kC);
  float* qb   = takeF((size_t)kT * kB * kC);
  float* attb = takeF((size_t)kT * kB * kC);
  int*   idxb = (int*)takeF((size_t)kB * kS * kK);  // neighbor indices
  _Float16* a16 = takeH((size_t)kS * kB * kF);      // GEMM A-operand f16 scratch
  float* hid  = bufA + (size_t)4 * 1024 * 1024;     // dec FFN hidden (after scores region)

  // f16 weight copies (converted once per call)
  _Float16* eWqkv = takeH((size_t)3 * kC * kC);
  _Float16* eWo   = takeH((size_t)kC * kC);
  _Float16* eW1   = takeH((size_t)kF * kC);
  _Float16* eW2   = takeH((size_t)kC * kF);
  _Float16* dW16[3][6];
  for (int d = 0; d < 3; ++d) {
    dW16[d][0] = takeH((size_t)3 * kC * kC);  // saW
    dW16[d][1] = takeH((size_t)kC * kC);      // saWo
    dW16[d][2] = takeH((size_t)3 * kC * kC);  // caW
    dW16[d][3] = takeH((size_t)kC * kC);      // caWo
    dW16[d][4] = takeH((size_t)kF * kC);      // W1
    dW16[d][5] = takeH((size_t)kC * kF);      // W2
  }

  // ---- convert weights f32 -> f16 (independent of everything else) ----
  cvt_launch(f(3),  eWqkv, (size_t)3 * kC * kC, stream);
  cvt_launch(f(5),  eWo,   (size_t)kC * kC,     stream);
  cvt_launch(f(7),  eW1,   (size_t)kF * kC,     stream);
  cvt_launch(f(9),  eW2,   (size_t)kC * kF,     stream);
  const int dbase[3] = {17, 37, 57};
  for (int d = 0; d < 3; ++d) {
    const int b0 = dbase[d];
    cvt_launch(f(b0 + 0),  dW16[d][0], (size_t)3 * kC * kC, stream);
    cvt_launch(f(b0 + 2),  dW16[d][1], (size_t)kC * kC,     stream);
    cvt_launch(f(b0 + 4),  dW16[d][2], (size_t)3 * kC * kC, stream);
    cvt_launch(f(b0 + 6),  dW16[d][3], (size_t)kC * kC,     stream);
    cvt_launch(f(b0 + 8),  dW16[d][4], (size_t)kF * kC,     stream);
    cvt_launch(f(b0 + 10), dW16[d][5], (size_t)kC * kF,     stream);
  }

  EncP E { eWqkv, eWo, eW1, eW2,
           f(4), f(6), f(8), f(10), f(11), f(12), f(13), f(14), f(15), f(16) };
  auto mkdec = [&](int di) {
    const int b0 = dbase[di];
    DecP D;
    D.saW = dW16[di][0]; D.saWo = dW16[di][1];
    D.caW = dW16[di][2]; D.caWo = dW16[di][3];
    D.W1  = dW16[di][4]; D.W2   = dW16[di][5];
    D.sab = f(b0+1);  D.sabo = f(b0+3);  D.cab = f(b0+5);  D.cabo = f(b0+7);
    D.b1  = f(b0+9);  D.b2   = f(b0+11);
    D.n1g = f(b0+12); D.n1b = f(b0+13); D.n2g = f(b0+14); D.n2b = f(b0+15);
    D.n3g = f(b0+16); D.n3b = f(b0+17); D.nfg = f(b0+18); D.nfb = f(b0+19);
    return D;
  };
  DecP Dn = mkdec(0), Dst = mkdec(1), Dt = mkdec(2);

  const int rS = kS * kB;  // 8192 token rows

  // ================= encoder =================
  {
    const int tot = kS * kB * kC;
    build_xs_k<<<(tot + 255) / 256, 256, 0, stream>>>(x, bufX);
  }
  knn_k<<<(kB * kS + 255) / 256, 256, 0, stream>>>(box, idxb);

  ln_launch(bufX, bufC, E.n1g, E.n1b, rS, stream);
  gemm_launch(bufC, a16, E.Wqkv, E.bqkv, nullptr, bufB, rS, 3 * kC, kC, 0, stream);
  enc_attn_k<<<(kB * kH * kS + 255) / 256, 256, 0, stream>>>(bufB, idxb, bufD);
  gemm_launch(bufD, a16, E.Wo, E.bo, bufX, bufX, rS, kC, kC, 0, stream);

  ln_launch(bufX, bufC, E.n2g, E.n2b, rS, stream);
  gemm_launch(bufC, a16, E.W1, E.b1, nullptr, bufA, rS, kF, kC, 1, stream);
  gemm_launch(bufA, a16, E.W2, E.b2, bufX, bufX, rS, kC, kF, 0, stream);
  ln_launch(bufX, bufX, E.nfg, E.nfb, rS, stream);          // xs final (in-place)

  mean_over_t_k<<<(kN * kB * kC + 255) / 256, 256, 0, stream>>>(bufX, outN);

  // ================= decoders =================
  // out = decoder(outN, xs, dec_n) -> d_out[0 : N*B*C]
  run_decoder(outN, kN, bufX, kS, Dn, (float*)d_out,
              t2b, qkvq, qb, bufB, bufA, attb, hid, a16, stream);

  // tgt = decoder(token, xs, dec_st) (kept in tokB)
  hipMemcpyAsync(tokB, token, (size_t)kT * kB * kC * sizeof(float),
                 hipMemcpyDeviceToDevice, stream);
  run_decoder(tokB, kT, bufX, kS, Dst, tokB,
              t2b, qkvq, qb, bufB, bufA, attb, hid, a16, stream);

  // tgt_t = decoder(mean(tgt), tgt, dec_t) -> d_out[N*B*C : ]
  mean_rows_k<<<(kB * kC + 255) / 256, 256, 0, stream>>>(tokB, tbuf, kT);
  run_decoder(tbuf, 1, tokB, kT, Dt, (float*)d_out + (size_t)kN * kB * kC,
              t2b, qkvq, qb, bufB, bufA, attb, hid, a16, stream);
}
